// TimeResampler_43868795961658
// MI455X (gfx1250) — matrix operand order repaired and checked by reading, not compiled, not run on hardware
//
#include <hip/hip_runtime.h>
#include <hip/hip_bf16.h>
#include <math.h>

// ---------------------------------------------------------------------------
// TimeResampler forward for MI455X (gfx1250, wave32, WMMA).
// All heavy GEMMs use v_wmma_f32_16x16x32_bf16 with fp32 accumulation.
// fp32 tensors live in global memory (L2-resident: 192MB), converted to bf16
// while staging tiles into LDS. GEMM is software-pipelined: next K-tile is
// prefetched from global into registers while WMMAs consume the current
// double-buffered LDS tile (one barrier per K-step).
// ---------------------------------------------------------------------------

typedef __bf16 bf16_t;
typedef bf16_t bf16x4  __attribute__((ext_vector_type(4)));
typedef bf16_t bf16x16 __attribute__((ext_vector_type(16)));
typedef float  f32x4   __attribute__((ext_vector_type(4)));
typedef float  f32x8   __attribute__((ext_vector_type(8)));

#define BATCH   32
#define SEQX    257
#define NQ      8
#define SEQKV   265     // 257 + 8
#define DIMSZ   1024
#define HEADS   16
#define DHEAD   64

union FragAB {
    uint4   u[2];
    bf16x16 v;
};

// ---------------------------------------------------------------------------
// Generic bf16-WMMA GEMM: C[M,N] = A[M,K] * B[K,N]  (+bias +rowadd, act, +resid)
// A,B fp32 row-major in global; converted to bf16 in LDS staging.
// Requires K % 32 == 0 and N % 128 == 0 (true for every call here).
// Block = 256 threads (8 waves). Tile 128x128, BK=32. Wave tile 32x64.
// Pipelined: register prefetch 1 K-step ahead + LDS double buffering.
// ---------------------------------------------------------------------------
__global__ __launch_bounds__(256)
void k_gemm(const float* __restrict__ A, const float* __restrict__ Bm,
            float* __restrict__ C, int M, int N, int K,
            const float* __restrict__ bias,
            const float* __restrict__ rowadd, int row_div,
            const float* __restrict__ resid, int act)
{
    constexpr int LP = 40;                       // LDS pitch (bf16), 80B rows
    __shared__ bf16_t As[2][128 * LP];           // A tile, row-major  [m][k]
    __shared__ bf16_t Bs[2][128 * LP];           // B tile, transposed [n][k]

    const int t  = threadIdx.x;
    const int m0 = blockIdx.y * 128;
    const int n0 = blockIdx.x * 128;

    const int w    = t >> 5;
    const int lane = t & 31;
    const int wm   = (w & 3) * 32;               // wave row base  (4 groups)
    const int wn   = (w >> 2) * 64;              // wave col base  (2 groups)
    const int lg   = lane >> 4;                  // lane K-half selector
    const int lm   = lane & 15;                  // m/n within 16

    f32x4 pa[4];                                 // global prefetch registers
    f32x4 pb[4];

    // issue global loads for K-tile `kk` into pa/pb
    auto gload = [&](int kk) {
        #pragma unroll
        for (int j = 0; j < 4; ++j) {
            int lin = t + j * 256;               // 0..1023
            int r   = lin >> 3;                  // 0..127
            int c4  = (lin & 7) << 2;            // 0,4,..,28
            f32x4 v = {0.f, 0.f, 0.f, 0.f};
            int gm = m0 + r;
            if (gm < M) v = *(const f32x4*)(A + (size_t)gm * K + kk + c4);
            pa[j] = v;
        }
        #pragma unroll
        for (int j = 0; j < 4; ++j) {
            int lin = t + j * 256;
            int kr  = lin >> 5;                  // 0..31
            int c4  = (lin & 31) << 2;           // 0..124
            pb[j] = *(const f32x4*)(Bm + (size_t)(kk + kr) * N + n0 + c4);
        }
    };

    // convert pa/pb to bf16 and store into LDS buffer `bsel`
    auto sstore = [&](int bsel) {
        #pragma unroll
        for (int j = 0; j < 4; ++j) {
            int lin = t + j * 256;
            int r   = lin >> 3;
            int c4  = (lin & 7) << 2;
            bf16x4 hv;
            hv[0] = (bf16_t)pa[j][0]; hv[1] = (bf16_t)pa[j][1];
            hv[2] = (bf16_t)pa[j][2]; hv[3] = (bf16_t)pa[j][3];
            *(bf16x4*)(&As[bsel][0] + r * LP + c4) = hv;
        }
        #pragma unroll
        for (int j = 0; j < 4; ++j) {
            int lin = t + j * 256;
            int kr  = lin >> 5;
            int c4  = (lin & 31) << 2;
            Bs[bsel][(c4 + 0) * LP + kr] = (bf16_t)pb[j][0];
            Bs[bsel][(c4 + 1) * LP + kr] = (bf16_t)pb[j][1];
            Bs[bsel][(c4 + 2) * LP + kr] = (bf16_t)pb[j][2];
            Bs[bsel][(c4 + 3) * LP + kr] = (bf16_t)pb[j][3];
        }
    };

    f32x8 acc[2][4] = {};

    // prologue: stage K-tile 0 into buffer 0
    gload(0);
    sstore(0);

    int buf = 0;
    for (int k0 = 0; k0 < K; k0 += 32) {
        __syncthreads();                         // staged tile `buf` visible

        const bool more = (k0 + 32) < K;
        if (more) gload(k0 + 32);                // prefetch next tile (regs)

        // ---- gather fragments per CDNA5 16-bit A/B VGPR layouts & compute
        FragAB fa[2];
        #pragma unroll
        for (int mt = 0; mt < 2; ++mt) {
            const bf16_t* pA = &As[buf][0] + (wm + mt * 16 + lm) * LP + lg * 8;
            fa[mt].u[0] = *(const uint4*)pA;         // K = lg*8 + 0..7
            fa[mt].u[1] = *(const uint4*)(pA + 16);  // K = lg*8 + 16..23
        }
        #pragma unroll
        for (int nt = 0; nt < 4; ++nt) {
            FragAB fb;
            const bf16_t* pB = &Bs[buf][0] + (wn + nt * 16 + lm) * LP + lg * 8;
            fb.u[0] = *(const uint4*)pB;             // K = lg*8 + 0..7
            fb.u[1] = *(const uint4*)(pB + 16);      // K = lg*8 + 16..23
            #pragma unroll
            for (int mt = 0; mt < 2; ++mt)
                acc[mt][nt] = __builtin_amdgcn_wmma_f32_16x16x32_bf16(
                    false, fa[mt].v, false, fb.v, (short)0, acc[mt][nt],
                    false, false);
        }

        if (more) sstore(buf ^ 1);               // write NEXT buffer (no race
                                                 // with readers of `buf`)
        buf ^= 1;
    }

    // ---- fused epilogue
    #pragma unroll
    for (int mt = 0; mt < 2; ++mt) {
        #pragma unroll
        for (int nt = 0; nt < 4; ++nt) {
            #pragma unroll
            for (int r = 0; r < 8; ++r) {
                int m = m0 + wm + mt * 16 + lg * 8 + r;
                int n = n0 + wn + nt * 16 + lm;
                if (m < M) {
                    float v = acc[mt][nt][r];
                    if (bias)   v += bias[n];
                    if (rowadd) v += rowadd[(size_t)(m / row_div) * N + n];
                    if (act == 1)   // exact GELU
                        v = 0.5f * v * (1.0f + erff(v * 0.70710678118654752f));
                    if (resid)  v += resid[(size_t)m * N + n];
                    C[(size_t)m * N + n] = v;
                }
            }
        }
    }
}

// ---------------------------------------------------------------------------
// LayerNorm over D=1024 with optional AdaLN modulation and row remapping.
// map: 0 identity, 1 x-rows -> kvin rows, 2 lat-rows -> kvin lat rows (+out2)
// ---------------------------------------------------------------------------
__global__ __launch_bounds__(256)
void k_ln(const float* __restrict__ in,
          const float* __restrict__ g, const float* __restrict__ b,
          const float* __restrict__ scale, const float* __restrict__ shift,
          int mod_div, int mod_stride,
          float* __restrict__ out, int map, float* __restrict__ out2)
{
    const int r = blockIdx.x, t = threadIdx.x;
    f32x4 x = *(const f32x4*)(in + (size_t)r * DIMSZ + t * 4);
    float s  = x[0] + x[1] + x[2] + x[3];
    float s2 = x[0]*x[0] + x[1]*x[1] + x[2]*x[2] + x[3]*x[3];
    #pragma unroll
    for (int off = 16; off; off >>= 1) {
        s  += __shfl_xor(s, off);
        s2 += __shfl_xor(s2, off);
    }
    __shared__ float rs[8], rs2[8], stat[2];
    if ((t & 31) == 0) { rs[t >> 5] = s; rs2[t >> 5] = s2; }
    __syncthreads();
    if (t == 0) {
        float a = 0.f, a2 = 0.f;
        for (int i = 0; i < 8; ++i) { a += rs[i]; a2 += rs2[i]; }
        float mean = a * (1.0f / DIMSZ);
        float var  = a2 * (1.0f / DIMSZ) - mean * mean;
        stat[0] = mean; stat[1] = rsqrtf(var + 1e-5f);
    }
    __syncthreads();
    const float mean = stat[0], inv = stat[1];

    size_t orow = (map == 1) ? (size_t)(r / SEQX) * SEQKV + (r % SEQX)
                : (map == 2) ? (size_t)(r / NQ) * SEQKV + SEQX + (r % NQ)
                : (size_t)r;
    const int bb = r / mod_div;
    #pragma unroll
    for (int e = 0; e < 4; ++e) {
        int d = t * 4 + e;
        float v = (x[e] - mean) * inv * g[d] + b[d];
        if (scale)
            v = v * (1.0f + scale[(size_t)bb * mod_stride + d])
                  + shift[(size_t)bb * mod_stride + d];
        out[orow * DIMSZ + d] = v;
        if (out2) out2[(size_t)r * DIMSZ + d] = v;
    }
}

// ---------------------------------------------------------------------------
// Attention: 1 block per (batch, head). q:[256,1024], kv:[8480,2048] fp32.
// 8 queries x 265 keys -> tiny; fp32 VALU softmax.
// ---------------------------------------------------------------------------
__global__ __launch_bounds__(256)
void k_attn(const float* __restrict__ q, const float* __restrict__ kv,
            float* __restrict__ out)
{
    const int b = blockIdx.x / HEADS;
    const int h = blockIdx.x % HEADS;
    const int t = threadIdx.x;

    __shared__ float qs[NQ][DHEAD];
    __shared__ float sc[NQ][272];

    #pragma unroll
    for (int idx = t; idx < NQ * DHEAD; idx += 256) {
        int l = idx / DHEAD, d = idx % DHEAD;
        qs[l][d] = q[(size_t)(b * NQ + l) * DIMSZ + h * DHEAD + d];
    }
    __syncthreads();

    for (int idx = t; idx < NQ * SEQKV; idx += 256) {
        int l = idx / SEQKV, n = idx % SEQKV;
        const float* kp = kv + (size_t)(b * SEQKV + n) * 2048 + h * DHEAD;
        float s = 0.f;
        #pragma unroll
        for (int d = 0; d < DHEAD; ++d) s += qs[l][d] * kp[d];
        sc[l][n] = s * 0.125f;     // (1/64^0.25)^2
    }
    __syncthreads();

    // softmax: wave w handles row w (8 waves, wave32)
    {
        int row = t >> 5, ln = t & 31;
        float mx = -3.4e38f;
        for (int n = ln; n < SEQKV; n += 32) mx = fmaxf(mx, sc[row][n]);
        #pragma unroll
        for (int off = 16; off; off >>= 1) mx = fmaxf(mx, __shfl_xor(mx, off));
        float sm = 0.f;
        for (int n = ln; n < SEQKV; n += 32) {
            float e = __expf(sc[row][n] - mx);
            sc[row][n] = e; sm += e;
        }
        #pragma unroll
        for (int off = 16; off; off >>= 1) sm += __shfl_xor(sm, off);
        float invs = 1.0f / sm;
        for (int n = ln; n < SEQKV; n += 32) sc[row][n] *= invs;
    }
    __syncthreads();

    #pragma unroll
    for (int idx = t; idx < NQ * DHEAD; idx += 256) {
        int l = idx / DHEAD, d = idx % DHEAD;
        float a = 0.f;
        for (int n = 0; n < SEQKV; ++n)
            a += sc[l][n] * kv[(size_t)(b * SEQKV + n) * 2048 + DIMSZ + h * DHEAD + d];
        out[(size_t)(b * NQ + l) * DIMSZ + h * DHEAD + d] = a;
    }
}

// ---------------------------------------------------------------------------
// Timestep embedding + time MLP stage 1: hidden = silu(emb @ w1 + b1)
// ---------------------------------------------------------------------------
__global__ __launch_bounds__(256)
void k_temb1(const int* __restrict__ ts, const float* __restrict__ w,
             const float* __restrict__ bias, float* __restrict__ hidden)
{
    const int b = blockIdx.x;
    const float tt = (float)ts[b];
    __shared__ float emb[320];
    for (int j = threadIdx.x; j < 320; j += 256) {
        int k = (j < 160) ? j : (j - 160);
        float f = __expf(-9.210340371976184f * (float)k / 160.0f);
        float a = tt * f;
        emb[j] = (j < 160) ? __cosf(a) : __sinf(a);   // flip_sin_to_cos
    }
    __syncthreads();
    for (int o = threadIdx.x; o < DIMSZ; o += 256) {
        float acc = bias[o];
        for (int k = 0; k < 320; ++k) acc += emb[k] * w[(size_t)k * DIMSZ + o];
        hidden[(size_t)b * DIMSZ + o] = acc / (1.0f + __expf(-acc));
    }
}

__global__ __launch_bounds__(256)
void k_temb2(const float* __restrict__ hidden, const float* __restrict__ w,
             const float* __restrict__ bias, float* __restrict__ temb)
{
    const int b = blockIdx.x;
    __shared__ float hs[DIMSZ];
    for (int k = threadIdx.x; k < DIMSZ; k += 256)
        hs[k] = hidden[(size_t)b * DIMSZ + k];
    __syncthreads();
    for (int o = threadIdx.x; o < DIMSZ; o += 256) {
        float acc = bias[o];
        for (int k = 0; k < DIMSZ; ++k) acc += hs[k] * w[(size_t)k * DIMSZ + o];
        temb[(size_t)b * DIMSZ + o] = acc;
    }
}

__global__ __launch_bounds__(256)
void k_silu(const float* __restrict__ in, float* __restrict__ out, int n)
{
    int i = blockIdx.x * 256 + threadIdx.x;
    if (i < n) { float v = in[i]; out[i] = v / (1.0f + __expf(-v)); }
}

__global__ __launch_bounds__(256)
void k_bcast(const float* __restrict__ latents, float* __restrict__ lat)
{
    const int r = blockIdx.x, t = threadIdx.x;
    f32x4 v = *(const f32x4*)(latents + (size_t)(r & (NQ - 1)) * DIMSZ + t * 4);
    *(f32x4*)(lat + (size_t)r * DIMSZ + t * 4) = v;
}

// ---------------------------------------------------------------------------
extern "C" void kernel_launch(void* const* d_in, const int* in_sizes, int n_in,
                              void* d_out, int out_size, void* d_ws, size_t ws_size,
                              hipStream_t stream)
{
    const float* x          = (const float*)d_in[0];
    const int*   tstep      = (const int*)  d_in[1];
    const float* latents    = (const float*)d_in[2];
    const float* proj_in_w  = (const float*)d_in[3];
    const float* proj_in_b  = (const float*)d_in[4];
    const float* time1_w    = (const float*)d_in[5];
    const float* time1_b    = (const float*)d_in[6];
    const float* time2_w    = (const float*)d_in[7];
    const float* time2_b    = (const float*)d_in[8];
    const float* n1_g       = (const float*)d_in[9];
    const float* n1_b       = (const float*)d_in[10];
    const float* n2_g       = (const float*)d_in[11];
    const float* n2_b       = (const float*)d_in[12];
    const float* wq         = (const float*)d_in[13];
    const float* wkv        = (const float*)d_in[14];
    const float* wo         = (const float*)d_in[15];
    const float* ff_g       = (const float*)d_in[16];
    const float* ff_b       = (const float*)d_in[17];
    const float* ff_w1      = (const float*)d_in[18];
    const float* ff_w2      = (const float*)d_in[19];
    const float* ada_w      = (const float*)d_in[20];
    const float* ada_b      = (const float*)d_in[21];
    const float* proj_out_w = (const float*)d_in[22];
    const float* proj_out_b = (const float*)d_in[23];
    const float* out_g      = (const float*)d_in[24];
    const float* out_b      = (const float*)d_in[25];

    // workspace (~142MB fp32)
    float* p = (float*)d_ws;
    auto take = [&](size_t n) { float* q = p; p += n; return q; };
    float* temb   = take((size_t)32 * 1024);
    float* hidden = take((size_t)32 * 1024);
    float* stemb  = take((size_t)32 * 1024);
    float* ada    = take((size_t)32 * 4096);
    float* h      = take((size_t)8224 * 1024);
    float* kvin   = take((size_t)8480 * 1024);
    float* kv     = take((size_t)8480 * 2048);
    float* qb     = take((size_t)256 * 1024);
    float* lnlq   = take((size_t)256 * 1024);
    float* attn   = take((size_t)256 * 1024);
    float* lat    = take((size_t)256 * 1024);
    float* yb     = take((size_t)256 * 1024);
    float* ffh    = take((size_t)256 * 4096);
    float* tout   = take((size_t)256 * 1024);

    auto gemm = [&](const float* A, const float* B, float* C,
                    int M, int N, int K,
                    const float* bias, const float* rowadd, int rowdiv,
                    const float* resid, int act) {
        dim3 g((N + 127) / 128, (M + 127) / 128);
        k_gemm<<<g, 256, 0, stream>>>(A, B, C, M, N, K, bias,
                                      rowadd, rowdiv, resid, act);
    };

    // temb = silu(emb @ w1 + b1) @ w2 + b2 ; stemb = silu(temb)
    k_temb1<<<32, 256, 0, stream>>>(tstep, time1_w, time1_b, hidden);
    k_temb2<<<32, 256, 0, stream>>>(hidden, time2_w, time2_b, temb);
    k_silu<<<128, 256, 0, stream>>>(temb, stemb, 32 * 1024);
    k_bcast<<<256, 256, 0, stream>>>(latents, lat);

    // h = x @ proj_in_w + b + temb[batch]
    gemm(x, proj_in_w, h, 8224, 1024, 768, proj_in_b, temb, SEQX, nullptr, 0);

    for (int i = 0; i < 8; ++i) {
        const float* wq_i    = wq    + (size_t)i * 1024 * 1024;
        const float* wkv_i   = wkv   + (size_t)i * 1024 * 2048;
        const float* wo_i    = wo    + (size_t)i * 1024 * 1024;
        const float* ffw1_i  = ff_w1 + (size_t)i * 1024 * 4096;
        const float* ffw2_i  = ff_w2 + (size_t)i * 4096 * 1024;
        const float* adaw_i  = ada_w + (size_t)i * 1024 * 4096;
        const float* adab_i  = ada_b + (size_t)i * 4096;
        const float* n1g_i = n1_g + (size_t)i * 1024;
        const float* n1b_i = n1_b + (size_t)i * 1024;
        const float* n2g_i = n2_g + (size_t)i * 1024;
        const float* n2b_i = n2_b + (size_t)i * 1024;
        const float* ffg_i = ff_g + (size_t)i * 1024;
        const float* ffb_i = ff_b + (size_t)i * 1024;

        // ada = silu(temb) @ ada_w + ada_b  -> [shift_msa|scale_msa|shift_mlp|scale_mlp]
        gemm(stemb, adaw_i, ada, 32, 4096, 1024, adab_i, nullptr, 1, nullptr, 0);
        // xn = LN(h) -> kvin rows [b*265 + 0..256]
        k_ln<<<8224, 256, 0, stream>>>(h, n1g_i, n1b_i, nullptr, nullptr,
                                       1, 0, kvin, 1, nullptr);
        // ln_lat = LN(lat)*(1+scale_msa)+shift_msa -> kvin rows [b*265+257..264] and lnlq
        k_ln<<<256, 256, 0, stream>>>(lat, n2g_i, n2b_i, ada + 1024, ada,
                                      NQ, 4096, kvin, 2, lnlq);
        // q = ln_lat @ wq
        gemm(lnlq, wq_i, qb, 256, 1024, 1024, nullptr, nullptr, 1, nullptr, 0);
        // kv = kvin @ wkv   (dominant GEMM: 8480x2048x1024)
        gemm(kvin, wkv_i, kv, 8480, 2048, 1024, nullptr, nullptr, 1, nullptr, 0);
        // attention
        k_attn<<<BATCH * HEADS, 256, 0, stream>>>(qb, kv, attn);
        // lat = attn @ wo + lat
        gemm(attn, wo_i, lat, 256, 1024, 1024, nullptr, nullptr, 1, lat, 0);
        // y = LN(lat)*(1+scale_mlp)+shift_mlp
        k_ln<<<256, 256, 0, stream>>>(lat, ffg_i, ffb_i, ada + 3072, ada + 2048,
                                      NQ, 4096, yb, 0, nullptr);
        // ffh = gelu(y @ ff_w1)
        gemm(yb, ffw1_i, ffh, 256, 4096, 1024, nullptr, nullptr, 1, nullptr, 1);
        // lat = ffh @ ff_w2 + lat
        gemm(ffh, ffw2_i, lat, 256, 1024, 4096, nullptr, nullptr, 1, lat, 0);
    }

    // out = LN(lat @ proj_out_w + b)
    gemm(lat, proj_out_w, tout, 256, 1024, 1024, proj_out_b, nullptr, 1, nullptr, 0);
    k_ln<<<256, 256, 0, stream>>>(tout, out_g, out_b, nullptr, nullptr,
                                  1, 0, (float*)d_out, 0, nullptr);
}
